// PoseFocusedIK_53274774339920
// MI455X (gfx1250) — matrix operand order repaired and checked by reading, not compiled, not run on hardware
//
#include <hip/hip_runtime.h>
#include <hip/hip_bf16.h>

// ---------------------------------------------------------------------------
// PoseFocusedIK forward, fused bf16-WMMA pipeline for gfx1250 (MI455X).
// GEMMs: v_wmma_f32_16x16x32_bf16, operands staged in LDS by the Tensor Data
// Mover (tensor_load_to_lds, double-buffered B tiles, TENSORcnt waits).
// ---------------------------------------------------------------------------

#define BATCH 65536

typedef __attribute__((ext_vector_type(16))) __bf16 v16bf;
typedef __attribute__((ext_vector_type(8)))  float  v8f;
typedef __attribute__((ext_vector_type(4)))  unsigned int u32x4;
typedef __attribute__((ext_vector_type(8)))  int          i32x8;
typedef __attribute__((ext_vector_type(4)))  int          i32x4;

union AFrag {
    unsigned int u[8];
    v16bf        v;
};

__device__ __forceinline__ unsigned short f2b(float f) {
    unsigned int u = __float_as_uint(f);
    unsigned int r = (u + 0x7FFFu + ((u >> 16) & 1u)) >> 16;
    return (unsigned short)r;
}
__device__ __forceinline__ float b2f(unsigned short h) {
    return __uint_as_float(((unsigned int)h) << 16);
}
__device__ __forceinline__ float gelu_exact(float x) {
    return 0.5f * x * (1.0f + erff(x * 0.70710678118654752f));
}
__device__ __forceinline__ float wave_sum(float s) {
#pragma unroll
    for (int m = 16; m >= 1; m >>= 1) s += __shfl_xor(s, m, 32);
    return s;
}
__device__ __forceinline__ unsigned lds_addr_of(const void* p) {
    // LDS aperture: flat addr low 32 bits are the LDS byte address
    return (unsigned)(unsigned long long)p;
}

// 2-D Tensor Data Mover load: tile (tile_d1 rows x tile_d0 elems) of a
// row-major [tensor_d1 x tensor_d0] bf16 tensor with row stride stride0
// (elements), packed contiguously into LDS at lds_addr.
__device__ __forceinline__ void tdm_load_2d(unsigned lds_addr,
                                            const void* gptr,
                                            unsigned tensor_d0,
                                            unsigned tensor_d1,
                                            unsigned tile_d0,
                                            unsigned tile_d1,
                                            unsigned long long stride0) {
    const unsigned long long ga = (unsigned long long)gptr;
    u32x4 g0;
    g0[0] = 1u;                                       // count=1, user mode
    g0[1] = lds_addr;                                 // LDS byte address
    g0[2] = (unsigned)(ga & 0xFFFFFFFFull);           // global_addr[31:0]
    g0[3] = (unsigned)((ga >> 32) & 0x01FFFFFFull)    // global_addr[56:32]
            | (2u << 30);                             // type = 2 ("image")
    i32x8 g1;
    g1[0] = (int)(1u << 16);                          // data_size = 2 bytes
    g1[1] = (int)((tensor_d0 & 0xFFFFu) << 16);       // tensor_dim0[15:0]
    g1[2] = (int)((tensor_d0 >> 16) | ((tensor_d1 & 0xFFFFu) << 16));
    g1[3] = (int)((tensor_d1 >> 16) | (tile_d0 << 16));
    g1[4] = (int)(tile_d1);                           // tile_dim1, tile_dim2=0
    g1[5] = (int)(stride0 & 0xFFFFFFFFull);           // dim0 stride low
    g1[6] = (int)((stride0 >> 32) & 0xFFFFull);       // dim0 stride high
    g1[7] = 0;
    i32x4 z4 = {0, 0, 0, 0};
    i32x8 z8 = {0, 0, 0, 0, 0, 0, 0, 0};
    // 6-arg variant (clang-23 / therock-10.0 headers)
    __builtin_amdgcn_tensor_load_to_lds(g0, g1, z4, z4, z8, 0);
}

// ---------------------------------------------------------------------------
// bf16 GEMM: out[B,256] = A[B,K](bf16) x W[K,256] + bias, W pre-transposed to
// WT[256,K] (n-major). Block: 256 threads = 8 waves; tile 64 rows x 256 cols.
// A slab (64xK) loaded once per block by TDM; B tiles (256x32) double-buffered
// by TDM. Inner loop is ds_load_b128 + v_wmma only.
// ---------------------------------------------------------------------------
template <int K>
__global__ __launch_bounds__(256) void gemm_tdm_n256(
    const unsigned short* __restrict__ A, const unsigned short* __restrict__ WT,
    const float* __restrict__ bias, float* __restrict__ outf, int fstride,
    unsigned short* __restrict__ outb, int bstride) {
    constexpr int NIT = K / 32;
    __shared__ __align__(16) unsigned short As[64 * K];
    __shared__ __align__(16) unsigned short Bs[2][256 * 32];

    const int tid  = threadIdx.x;
    const int lane = tid & 31;
    const int wave = tid >> 5;
    const int wm   = wave & 3;
    const int wn   = wave >> 2;
    const long rowBase = (long)blockIdx.x * 64;

    if (wave == 0) {
        // Whole A slab: 64 rows x K (row-contiguous)
        tdm_load_2d(lds_addr_of(As), A + rowBase * K, (unsigned)K, 64u,
                    (unsigned)K, 64u, (unsigned long long)K);
        // First B tile: 256 rows x 32 cols of WT (row stride K)
        tdm_load_2d(lds_addr_of(Bs[0]), WT, (unsigned)K, 256u, 32u, 256u,
                    (unsigned long long)K);
    }

    v8f acc[8];
#pragma unroll
    for (int i = 0; i < 8; i++)
#pragma unroll
        for (int r = 0; r < 8; r++) acc[i][r] = 0.0f;

    for (int i = 0; i < NIT; i++) {
        const int kc = i * 32;
        if (wave == 0) {
            if (i + 1 < NIT) {
                // prefetch next B tile into the other buffer, keep 1 in flight
                tdm_load_2d(lds_addr_of(Bs[(i + 1) & 1]), WT + (i + 1) * 32,
                            (unsigned)K, 256u, 32u, 256u,
                            (unsigned long long)K);
                __builtin_amdgcn_s_wait_tensorcnt((short)1);
            } else {
                __builtin_amdgcn_s_wait_tensorcnt((short)0);
            }
        }
        __syncthreads();  // TDM data visible to all waves

        // A fragment per ISA 16-bit A 16x32 layout:
        // lanes0-15: M=lane, K pairs {0..7,16..23}; lanes16-31: K {8..15,24..31}
        AFrag af;
        {
            const int m  = wm * 16 + (lane & 15);
            const int kb = (lane < 16) ? 0 : 8;
            const unsigned short* ar = &As[m * K + kc];
#pragma unroll
            for (int p = 0; p < 4; p++) {
                af.u[p]     = *(const unsigned int*)&ar[kb + 2 * p];
                af.u[4 + p] = *(const unsigned int*)&ar[16 + kb + 2 * p];
            }
        }
        // B fragments per ISA 16-bit B 32x16 layout:
        // lanes0-15: N=lane, K {0..7,16..23}; lanes16-31: N=lane-16, K {8..15,24..31}
        const unsigned short* Bcur = Bs[i & 1];
        const int ncol0 = lane & 15;
        const int ks    = (lane < 16) ? 0 : 8;
#pragma unroll
        for (int nt = 0; nt < 8; nt++) {
            AFrag bf;
            const unsigned short* br =
                &Bcur[(wn * 128 + nt * 16 + ncol0) * 32 + ks];
#pragma unroll
            for (int p = 0; p < 8; p++) bf.u[p] = *(const unsigned int*)&br[2 * (p & 3) + 16 * (p >> 2)];
            acc[nt] = __builtin_amdgcn_wmma_f32_16x16x32_bf16(
                false, af.v, false, bf.v, (short)0, acc[nt], false, false);
        }
        __syncthreads();  // protect buffer before next TDM overwrite
    }

    // D layout: VGPR r -> M = wm*16 + r + (lane>=16 ? 8 : 0)
    const int mr = wm * 16 + ((lane >= 16) ? 8 : 0);
    const int nc = wn * 128 + (lane & 15);
#pragma unroll
    for (int nt = 0; nt < 8; nt++) {
#pragma unroll
        for (int r = 0; r < 8; r++) {
            const int n = nc + nt * 16;
            float v = acc[nt][r];
            if (bias) v += bias[n];
            const long row = rowBase + mr + r;
            if (outf) outf[row * (long)fstride + n] = v;
            if (outb) outb[row * (long)bstride + n] = f2b(v);
        }
    }
}

// ---------------------------------------------------------------------------
// History compressor + history encoder (wave per row)
// ---------------------------------------------------------------------------
__global__ __launch_bounds__(256) void preproc_kernel(
    const float* __restrict__ hf, const float* __restrict__ w1,
    const float* __restrict__ b1, const float* __restrict__ w2,
    const float* __restrict__ b2, const float* __restrict__ wh,
    const float* __restrict__ bh, const float* __restrict__ lng,
    const float* __restrict__ lnb, float* __restrict__ hist_out,
    unsigned short* __restrict__ hb) {
    const long row  = (long)blockIdx.x * 8 + (threadIdx.x >> 5);
    const int  lane = threadIdx.x & 31;
    const float* fr = hf + row * 70;

    float xj = 0.0f;
    if (lane < 7) {
#pragma unroll
        for (int f = 0; f < 10; f++) xj += fr[f * 7 + lane];
        xj *= 0.1f;
    }
    float x[7];
#pragma unroll
    for (int j = 0; j < 7; j++) x[j] = __shfl(xj, j, 32);

    float a1[8];
#pragma unroll
    for (int o = 0; o < 8; o++) {
        const int n = lane * 8 + o;
        float s = b1[n];
#pragma unroll
        for (int j = 0; j < 7; j++) s += x[j] * w1[j * 256 + n];
        a1[o] = fmaxf(s, 0.0f);
    }
    float cmp[7];
#pragma unroll
    for (int j = 0; j < 7; j++) {
        float s = 0.0f;
#pragma unroll
        for (int o = 0; o < 8; o++) s += a1[o] * w2[(lane * 8 + o) * 7 + j];
        s = wave_sum(s) + b2[j];
        const float attn = 1.0f / (1.0f + expf(-s));
        cmp[j] = x[j] * attn;
    }
    float h[8];
    float s = 0.0f;
#pragma unroll
    for (int o = 0; o < 8; o++) {
        const int n = lane * 8 + o;
        float t = bh[n];
#pragma unroll
        for (int j = 0; j < 7; j++) t += cmp[j] * wh[j * 256 + n];
        h[o] = t;
        s += t;
    }
    const float mean = wave_sum(s) * (1.0f / 256.0f);
    float q = 0.0f;
#pragma unroll
    for (int o = 0; o < 8; o++) { const float d = h[o] - mean; q += d * d; }
    const float rstd = rsqrtf(wave_sum(q) * (1.0f / 256.0f) + 1e-5f);
#pragma unroll
    for (int o = 0; o < 8; o++) {
        const int n = lane * 8 + o;
        float y = lng[n] * (h[o] - mean) * rstd + lnb[n];
        y = gelu_exact(y);
        hist_out[row * 256 + n] = y;
        hb[row * 256 + n] = f2b(y);
    }
}

// Small encoder: din(3/4) -> 128, LN, GELU, bf16 out (wave per row)
__global__ __launch_bounds__(256) void enc_small128(
    const float* __restrict__ in, int din, const float* __restrict__ W,
    const float* __restrict__ bias, const float* __restrict__ g,
    const float* __restrict__ bln, unsigned short* __restrict__ outb) {
    const long row  = (long)blockIdx.x * 8 + (threadIdx.x >> 5);
    const int  lane = threadIdx.x & 31;
    float xin[4];
    for (int j = 0; j < din; j++) xin[j] = in[row * din + j];
    float v[4];
    float s = 0.0f;
#pragma unroll
    for (int o = 0; o < 4; o++) {
        const int n = lane * 4 + o;
        float t = bias[n];
        for (int j = 0; j < din; j++) t += xin[j] * W[j * 128 + n];
        v[o] = t;
        s += t;
    }
    const float mean = wave_sum(s) * (1.0f / 128.0f);
    float q = 0.0f;
#pragma unroll
    for (int o = 0; o < 4; o++) { const float d = v[o] - mean; q += d * d; }
    const float rstd = rsqrtf(wave_sum(q) * (1.0f / 128.0f) + 1e-5f);
#pragma unroll
    for (int o = 0; o < 4; o++) {
        const int n = lane * 4 + o;
        float y = g[n] * (v[o] - mean) * rstd + bln[n];
        outb[row * 128 + n] = f2b(gelu_exact(y));
    }
}

// LayerNorm(d=256) + GELU, optional f32/bf16 outputs with strides
__global__ __launch_bounds__(256) void ln_gelu_256(
    const float* __restrict__ in, const float* __restrict__ g,
    const float* __restrict__ b, float* __restrict__ outf, int fstride,
    unsigned short* __restrict__ outb, int bstride) {
    const long row  = (long)blockIdx.x * 8 + (threadIdx.x >> 5);
    const int  lane = threadIdx.x & 31;
    const float* x = in + row * 256 + lane * 8;
    float v[8];
    float s = 0.0f;
#pragma unroll
    for (int o = 0; o < 8; o++) { v[o] = x[o]; s += v[o]; }
    const float mean = wave_sum(s) * (1.0f / 256.0f);
    float q = 0.0f;
#pragma unroll
    for (int o = 0; o < 8; o++) { const float d = v[o] - mean; q += d * d; }
    const float rstd = rsqrtf(wave_sum(q) * (1.0f / 256.0f) + 1e-5f);
#pragma unroll
    for (int o = 0; o < 8; o++) {
        const int n = lane * 8 + o;
        float y = g[n] * (v[o] - mean) * rstd + b[n];
        y = gelu_exact(y);
        if (outf) outf[row * (long)fstride + n] = y;
        if (outb) outb[row * (long)bstride + n] = f2b(y);
    }
}

// Adaptive fusion gating -> builds [wh || wp] bf16 concat
__global__ __launch_bounds__(256) void whwp_kernel(
    const float* __restrict__ gatepre, const float* __restrict__ hist,
    const float* __restrict__ pose, unsigned short* __restrict__ cat) {
    const long i = (long)blockIdx.x * 256 + threadIdx.x;
    const long row = i >> 8;
    const int  n   = (int)(i & 255);
    const float gs = 1.0f / (1.0f + expf(-gatepre[i]));
    cat[row * 512 + n]       = f2b(hist[i] * gs * 0.3f);
    cat[row * 512 + 256 + n] = f2b(pose[i] * (1.0f - gs) * 1.3f);
}

__global__ __launch_bounds__(256) void pack_lower(
    const unsigned short* __restrict__ src, unsigned short* __restrict__ cat) {
    const long i = (long)blockIdx.x * 256 + threadIdx.x;
    const long row = i >> 8;
    const int  n   = (int)(i & 255);
    cat[row * 512 + n] = src[i];
}

__global__ __launch_bounds__(256) void resadd_bf16(
    unsigned short* __restrict__ g, const float* __restrict__ add) {
    const long i = (long)blockIdx.x * 256 + threadIdx.x;
    g[i] = f2b(b2f(g[i]) + add[i]);
}

// f32 [K,256] -> bf16 transposed [256,K] (reads coalesced; once per launch)
__global__ __launch_bounds__(256) void cvt_transpose_bf16(
    const float* __restrict__ in, unsigned short* __restrict__ out, int K) {
    const int i = blockIdx.x * 256 + threadIdx.x;
    if (i < K * 256) {
        const int k = i >> 8;
        const int n = i & 255;
        out[n * K + k] = f2b(in[i]);
    }
}

// Output heads + pose reconstruction (wave per row)
__global__ __launch_bounds__(256) void heads_recon(
    const unsigned short* __restrict__ G1, const unsigned short* __restrict__ G2,
    const unsigned short* __restrict__ G3, const unsigned short* __restrict__ G4,
    const float* __restrict__ wsh, const float* __restrict__ bsh,
    const float* __restrict__ wel, const float* __restrict__ bel,
    const float* __restrict__ wfa, const float* __restrict__ bfa,
    const float* __restrict__ wwr, const float* __restrict__ bwr,
    const float* __restrict__ wr1, const float* __restrict__ br1,
    const float* __restrict__ wr2, const float* __restrict__ br2,
    float* __restrict__ pred_out, float* __restrict__ recon_out) {
    const long row  = (long)blockIdx.x * 8 + (threadIdx.x >> 5);
    const int  lane = threadIdx.x & 31;
    float a[8], bb[8], c[8], d[8];
#pragma unroll
    for (int o = 0; o < 8; o++) {
        const int n = lane * 8 + o;
        a[o]  = b2f(G1[row * 256 + n]);
        bb[o] = b2f(G2[row * 256 + n]);
        c[o]  = b2f(G3[row * 256 + n]);
        d[o]  = b2f(G4[row * 256 + n]);
    }
    float pred[7];
#pragma unroll
    for (int t = 0; t < 3; t++) {
        float s = 0.0f;
#pragma unroll
        for (int o = 0; o < 8; o++) s += a[o] * wsh[(lane * 8 + o) * 3 + t];
        pred[t] = wave_sum(s) + bsh[t];
    }
    {
        float s = 0.0f;
#pragma unroll
        for (int o = 0; o < 8; o++) s += bb[o] * wel[lane * 8 + o];
        pred[3] = wave_sum(s) + bel[0];
    }
    {
        float s = 0.0f;
#pragma unroll
        for (int o = 0; o < 8; o++) s += c[o] * wfa[lane * 8 + o];
        pred[4] = wave_sum(s) + bfa[0];
    }
#pragma unroll
    for (int t = 0; t < 2; t++) {
        float s = 0.0f;
#pragma unroll
        for (int o = 0; o < 8; o++) s += d[o] * wwr[(lane * 8 + o) * 2 + t];
        pred[5 + t] = wave_sum(s) + bwr[t];
    }
    if (lane == 0) {
#pragma unroll
        for (int j = 0; j < 7; j++) pred_out[row * 7 + j] = pred[j];
    }
    float r[8];
#pragma unroll
    for (int o = 0; o < 8; o++) {
        const int n = lane * 8 + o;
        float s = br1[n];
#pragma unroll
        for (int j = 0; j < 7; j++) s += pred[j] * wr1[j * 256 + n];
        r[o] = fmaxf(s, 0.0f);
    }
    float rec[7];
#pragma unroll
    for (int j = 0; j < 7; j++) {
        float s = 0.0f;
#pragma unroll
        for (int o = 0; o < 8; o++) s += r[o] * wr2[(lane * 8 + o) * 7 + j];
        rec[j] = wave_sum(s) + br2[j];
    }
    if (lane == 0) {
#pragma unroll
        for (int j = 0; j < 7; j++) recon_out[row * 7 + j] = rec[j];
    }
}

// ---------------------------------------------------------------------------
// Host-side orchestration
// ---------------------------------------------------------------------------
extern "C" void kernel_launch(void* const* d_in, const int* in_sizes, int n_in,
                              void* d_out, int out_size, void* d_ws,
                              size_t ws_size, hipStream_t stream) {
    (void)in_sizes; (void)n_in; (void)out_size; (void)ws_size;
    const long B = BATCH;
    auto P = [&](int i) { return (const float*)d_in[i]; };

    // Workspace layout
    char* ws = (char*)d_ws;
    size_t off = 0;
    float* F0 = (float*)(ws + off);                    off += (size_t)B * 256 * 4;
    unsigned short* CAT = (unsigned short*)(ws + off); off += (size_t)B * 512 * 2;
    unsigned short* X1 = (unsigned short*)(ws + off);  off += (size_t)B * 256 * 2;
    unsigned short* X2 = (unsigned short*)(ws + off);  off += (size_t)B * 128 * 2;
    unsigned short* HB = (unsigned short*)(ws + off);  off += (size_t)B * 256 * 2;
    unsigned short* G1 = (unsigned short*)(ws + off);  off += (size_t)B * 256 * 2;
    unsigned short* G2 = (unsigned short*)(ws + off);  off += (size_t)B * 256 * 2;
    unsigned short* G3 = (unsigned short*)(ws + off);  off += (size_t)B * 256 * 2;
    unsigned short* G4 = (unsigned short*)(ws + off);  off += (size_t)B * 256 * 2;
    unsigned short* WB = (unsigned short*)(ws + off);
    unsigned short* wb_pos2 = WB;                       // WT 256x128
    unsigned short* wb_pos3 = wb_pos2 + 128 * 256;      // WT 256x256
    unsigned short* wb_ori2 = wb_pos3 + 256 * 256;      // WT 256x128
    unsigned short* wb_ori3 = wb_ori2 + 128 * 256;      // WT 256x256
    unsigned short* wb_pf   = wb_ori3 + 256 * 256;      // WT 256x512
    unsigned short* wb_gate = wb_pf   + 512 * 256;      // WT 256x512
    unsigned short* wb_fuse = wb_gate + 512 * 256;      // WT 256x512
    unsigned short* wb_ge   = wb_fuse + 512 * 256;      // 4 x WT 256x256
    unsigned short* wb_msg  = wb_ge   + 4 * 256 * 256;  // 6 x WT 256x256

    float* out_pred  = (float*)d_out;
    float* out_recon = out_pred + B * 7;
    float* out_hist  = out_pred + B * 14;
    float* out_pose  = out_hist + B * 256;

    const int ROWS8 = (int)(B / 8);   // wave-per-row kernels
    const int GB64  = (int)(B / 64);  // GEMM blocks
    const int EW    = (int)(B);       // B*256/256 elementwise blocks

#define CVT_T(src, dst, K) \
    cvt_transpose_bf16<<<((K) * 256 + 255) / 256, 256, 0, stream>>>(src, dst, K)

    // Weight f32 -> bf16 + transpose to n-major [256,K]
    CVT_T(P(15), wb_pos2, 128);
    CVT_T(P(19), wb_pos3, 256);
    CVT_T(P(25), wb_ori2, 128);
    CVT_T(P(29), wb_ori3, 256);
    CVT_T(P(31), wb_pf,   512);
    CVT_T(P(35), wb_gate, 512);
    CVT_T(P(37), wb_fuse, 512);
    CVT_T(P(41), wb_ge + 0 * 65536, 256);
    CVT_T(P(43), wb_ge + 1 * 65536, 256);
    CVT_T(P(45), wb_ge + 2 * 65536, 256);
    CVT_T(P(47), wb_ge + 3 * 65536, 256);
    CVT_T(P(49), wb_msg + 0 * 65536, 256);
    CVT_T(P(51), wb_msg + 1 * 65536, 256);
    CVT_T(P(53), wb_msg + 2 * 65536, 256);
    CVT_T(P(55), wb_msg + 3 * 65536, 256);
    CVT_T(P(57), wb_msg + 4 * 65536, 256);
    CVT_T(P(59), wb_msg + 5 * 65536, 256);

    // History compressor + history encoder -> out_hist(f32), HB(bf16)
    preproc_kernel<<<ROWS8, 256, 0, stream>>>(
        P(0), P(3), P(4), P(5), P(6), P(7), P(8), P(9), P(10), out_hist, HB);

    // Position branch
    enc_small128<<<ROWS8, 256, 0, stream>>>(P(1), 3, P(11), P(12), P(13), P(14), X2);
    gemm_tdm_n256<128><<<GB64, 256, 0, stream>>>(X2, wb_pos2, P(16), F0, 256, nullptr, 0);
    ln_gelu_256<<<ROWS8, 256, 0, stream>>>(F0, P(17), P(18), nullptr, 0, X1, 256);
    gemm_tdm_n256<256><<<GB64, 256, 0, stream>>>(X1, wb_pos3, P(20), nullptr, 0, CAT, 512);

    // Orientation branch
    enc_small128<<<ROWS8, 256, 0, stream>>>(P(2), 4, P(21), P(22), P(23), P(24), X2);
    gemm_tdm_n256<128><<<GB64, 256, 0, stream>>>(X2, wb_ori2, P(26), F0, 256, nullptr, 0);
    ln_gelu_256<<<ROWS8, 256, 0, stream>>>(F0, P(27), P(28), nullptr, 0, X1, 256);
    gemm_tdm_n256<256><<<GB64, 256, 0, stream>>>(X1, wb_ori3, P(30), nullptr, 0, CAT + 256, 512);

    // Pose fusion: [pos||ori] -> pose_feat (f32 out + bf16 into CAT upper half)
    gemm_tdm_n256<512><<<GB64, 256, 0, stream>>>(CAT, wb_pf, P(32), F0, 256, nullptr, 0);
    ln_gelu_256<<<ROWS8, 256, 0, stream>>>(F0, P(33), P(34), out_pose, 256, CAT + 256, 512);

    // Adaptive fusion: CAT = [hist||pose] -> gate -> CAT = [wh||wp] -> fused(HB)
    pack_lower<<<EW, 256, 0, stream>>>(HB, CAT);
    gemm_tdm_n256<512><<<GB64, 256, 0, stream>>>(CAT, wb_gate, P(36), F0, 256, nullptr, 0);
    whwp_kernel<<<EW, 256, 0, stream>>>(F0, out_hist, out_pose, CAT);
    gemm_tdm_n256<512><<<GB64, 256, 0, stream>>>(CAT, wb_fuse, P(38), F0, 256, nullptr, 0);
    ln_gelu_256<<<ROWS8, 256, 0, stream>>>(F0, P(39), P(40), nullptr, 0, HB, 256);

    // Joint-group encoders
    gemm_tdm_n256<256><<<GB64, 256, 0, stream>>>(HB, wb_ge + 0 * 65536, P(42), nullptr, 0, G1, 256);
    gemm_tdm_n256<256><<<GB64, 256, 0, stream>>>(HB, wb_ge + 1 * 65536, P(44), nullptr, 0, G2, 256);
    gemm_tdm_n256<256><<<GB64, 256, 0, stream>>>(HB, wb_ge + 2 * 65536, P(46), nullptr, 0, G3, 256);
    gemm_tdm_n256<256><<<GB64, 256, 0, stream>>>(HB, wb_ge + 3 * 65536, P(48), nullptr, 0, G4, 256);

    // Message passing x2: shoulder->elbow->forearm->wrist with residuals
    for (int l = 0; l < 2; l++) {
        const int base = l * 3;
        gemm_tdm_n256<256><<<GB64, 256, 0, stream>>>(
            G1, wb_msg + (base + 0) * 65536, P(49 + l * 6 + 1), F0, 256, nullptr, 0);
        resadd_bf16<<<EW, 256, 0, stream>>>(G2, F0);
        gemm_tdm_n256<256><<<GB64, 256, 0, stream>>>(
            G2, wb_msg + (base + 1) * 65536, P(49 + l * 6 + 3), F0, 256, nullptr, 0);
        resadd_bf16<<<EW, 256, 0, stream>>>(G3, F0);
        gemm_tdm_n256<256><<<GB64, 256, 0, stream>>>(
            G3, wb_msg + (base + 2) * 65536, P(49 + l * 6 + 5), F0, 256, nullptr, 0);
        resadd_bf16<<<EW, 256, 0, stream>>>(G4, F0);
    }

    // Heads + reconstruction
    heads_recon<<<ROWS8, 256, 0, stream>>>(
        G1, G2, G3, G4, P(61), P(62), P(63), P(64), P(65), P(66), P(67), P(68),
        P(69), P(70), P(71), P(72), out_pred, out_recon);
#undef CVT_T
}
